// GCNLSTM_89515708383436
// MI455X (gfx1250) — matrix operand order repaired and checked by reading, not compiled, not run on hardware
//
#include <hip/hip_runtime.h>
#include <stdint.h>

// ---------------- problem constants (match reference) ----------------
constexpr int B_  = 2, T_ = 6, CIN_ = 64, N_ = 8192, E_ = 131072, H_ = 128;
constexpr int M_  = B_ * N_;        // 16384 GEMM rows (batch*nodes)
constexpr int MT_ = M_ / 16;        // 1024 row tiles
constexpr int NCOL_ = 4 * H_;       // 512 gate columns (i,f,c~,o)
constexpr int NT_ = NCOL_ / 16;     // 32 column tiles
constexpr int KT0_ = (3 * CIN_ + 3 * H_) / 32;  // 18 K-tiles (layer0: 192+384)
constexpr int KT1_ = (3 * H_ + 3 * H_) / 32;    // 24 K-tiles (layer1: 384+384)

typedef __attribute__((ext_vector_type(16))) _Float16 v16h;
typedef __attribute__((ext_vector_type(8)))  float    v8f;
typedef __attribute__((ext_vector_type(4)))  float    v4f;
typedef __attribute__((ext_vector_type(4)))  unsigned int u32x4;
typedef __attribute__((ext_vector_type(8)))  int      i32x8;
typedef __attribute__((ext_vector_type(4)))  int      i32x4;

__device__ __forceinline__ float sigm(float x) { return 1.0f / (1.0f + __expf(-x)); }

// ---------------- utility: zero fill ----------------
__global__ void k_zero(float* __restrict__ p, int n) {
  int i = blockIdx.x * blockDim.x + threadIdx.x;
  if (i < n) p[i] = 0.0f;
}

// ---------------- graph normalization ----------------
__global__ void k_deg(const int* __restrict__ ei, const float* __restrict__ ea,
                      float* __restrict__ deg) {
  int e = blockIdx.x * blockDim.x + threadIdx.x;
  if (e < E_) atomicAdd(&deg[ei[e]], ea[e]);
}
__global__ void k_dinv(float* __restrict__ deg) {
  int n = blockIdx.x * blockDim.x + threadIdx.x;
  if (n < N_) { float d = deg[n]; deg[n] = (d > 0.0f) ? rsqrtf(d) : 0.0f; }
}
__global__ void k_wn(const int* __restrict__ ei, const float* __restrict__ ea,
                     const float* __restrict__ dinv, float* __restrict__ wn) {
  int e = blockIdx.x * blockDim.x + threadIdx.x;
  if (e < E_) wn[e] = -dinv[ei[e]] * ea[e] * dinv[ei[E_ + e]];
}

// ---------------- input slice + normalize:  x_t[m][c], c==1 zeroed ----------------
__global__ void k_xslice(const float* __restrict__ in, const float* __restrict__ mu,
                         const float* __restrict__ ls, float* __restrict__ xbuf, int t) {
  int tid = blockIdx.x * blockDim.x + threadIdx.x;
  if (tid >= M_ * CIN_) return;
  int m = tid >> 6, c = tid & 63;
  int b = m >> 13, n = m & (N_ - 1);
  float v = 0.0f;
  if (c != 1) {
    size_t pi = ((size_t)(b * T_ + t) * CIN_ + c) * N_ + n;
    size_t pc = (size_t)c * N_ + n;
    v = (in[pi] - mu[pc]) / (__expf(ls[pc]) + 1e-6f);
  }
  xbuf[tid] = v;
}

// ---------------- sparse propagation: o[:,dst] += wn * z[:,src] ----------------
__global__ void k_prop(const float* __restrict__ z, float* __restrict__ o,
                       const int* __restrict__ ei, const float* __restrict__ wn, int F) {
  int tid = blockIdx.x * blockDim.x + threadIdx.x;
  int q = F >> 2;
  if (tid >= E_ * q) return;
  int e = tid / q, f4 = tid - e * q;
  int s = ei[e], d = ei[E_ + e];
  float w = wn[e];
#pragma unroll
  for (int b = 0; b < B_; ++b) {
    const v4f z4 = *(const v4f*)(z + ((size_t)(b * N_ + s)) * F + f4 * 4);
    float* op = o + ((size_t)(b * N_ + d)) * F + f4 * 4;
    atomicAdd(op + 0, w * z4.x);
    atomicAdd(op + 1, w * z4.y);
    atomicAdd(op + 2, w * z4.z);
    atomicAdd(op + 3, w * z4.w);
  }
}

// ---------------- pack A = [x|P1|2P2-x | h|P1h|2P2h-h] into WMMA A-fragments ----
// Fragment layout per ISA 16-bit A 16x32 table:
//   lane L, elem e: v=e/2,hb=e&1; K = (v&4?16:0) + (L&16?8:0) + (v&3)*2 + hb; M=L&15
// Stored lane-contiguous: Af[((mt*KT+kt)*32+lane)*16+e]  (f16)
__global__ void k_pack(const float* __restrict__ X, const float* __restrict__ P1,
                       const float* __restrict__ P2, int Fx,
                       const float* __restrict__ Hs, const float* __restrict__ P1h,
                       const float* __restrict__ P2h,
                       _Float16* __restrict__ Af, int KT) {
  int tid = blockIdx.x * blockDim.x + threadIdx.x;
  if (tid >= MT_ * KT * 512) return;
  int e = tid & 15;
  int lane = (tid >> 4) & 31;
  int rest = tid >> 9;
  int kt = rest % KT;
  int mt = rest / KT;
  int v = e >> 1, hb = e & 1;
  int K = kt * 32 + ((v & 4) ? 16 : 0) + ((lane & 16) ? 8 : 0) + ((v & 3) << 1) + hb;
  int m = (mt << 4) + (lane & 15);
  int Kx = 3 * Fx;
  float val;
  if (K < Kx) {
    int sel = K / Fx, f = K - sel * Fx;
    size_t o = (size_t)m * Fx + f;
    val = (sel == 0) ? X[o] : (sel == 1) ? P1[o] : 2.0f * P2[o] - X[o];
  } else {
    int K2 = K - Kx;
    int sel = K2 / H_, f = K2 - sel * H_;
    size_t o = (size_t)m * H_ + f;
    val = (sel == 0) ? Hs[o] : (sel == 1) ? P1h[o] : 2.0f * P2h[o] - Hs[o];
  }
  Af[tid] = (_Float16)val;
}

// ---------------- weight prep: fp32 [4,3,Fin,H] pair -> f16 B-fragments --------
// B layout (32x16): lane L holds N=L&15; elem e: K = (e&8?16:0)+(L&16?8:0)+(e&7)
__global__ void k_prep_w(const float* __restrict__ Wx, const float* __restrict__ Wh,
                         int Fx, _Float16* __restrict__ Wf, int KT) {
  int tid = blockIdx.x * blockDim.x + threadIdx.x;
  if (tid >= KT * NT_ * 512) return;
  int e = tid & 15;
  int lane = (tid >> 4) & 31;
  int rest = tid >> 9;
  int nt = rest % NT_;
  int kt = rest / NT_;
  int K = kt * 32 + ((e & 8) ? 16 : 0) + ((lane & 16) ? 8 : 0) + (e & 7);
  int n = (nt << 4) + (lane & 15);
  int g = n >> 7, h = n & 127;
  int Kx = 3 * Fx;
  float val;
  if (K < Kx) {
    int k = K / Fx, f = K - k * Fx;
    val = Wx[(((size_t)(g * 3 + k)) * Fx + f) * H_ + h];
  } else {
    int K2 = K - Kx;
    int k = K2 / H_, f = K2 - k * H_;
    val = Wh[(((size_t)(g * 3 + k)) * H_ + f) * H_ + h];
  }
  Wf[tid] = (_Float16)val;
}

__global__ void k_prep_bias(const float* __restrict__ bx, const float* __restrict__ bh,
                            const float* __restrict__ bg, float* __restrict__ bs) {
  int i = blockIdx.x * blockDim.x + threadIdx.x;
  if (i < NCOL_) bs[i] = bx[i] + bh[i] + bg[i];
}

// ---------------- WMMA GEMM: pre[M,512] = A[M,KT*32] x W[KT*32,512] -----------
// Block = 8 waves, 32 rows (2 M-tiles) per block for 2x weight reuse from L2.
// A slab (2*KT KB) staged to LDS: first half via Tensor Data Mover (TENSORcnt),
// second half via global_load_async_to_lds_b128 (ASYNCcnt).
__global__ __launch_bounds__(256) void k_gemm(const _Float16* __restrict__ Af,
                                              const _Float16* __restrict__ Wf,
                                              float* __restrict__ Cout, int KT) {
  __shared__ _Float16 shA[2 * KT1_ * 512];  // max: 2 slabs * 24 KB = 48 KB
  const int tid = threadIdx.x;
  const int lane = tid & 31;
  const int wv = tid >> 5;
  const int bt = blockIdx.x;            // 512 blocks, rows [bt*32, bt*32+32)

  const int slabHalf = KT * 512;        // f16 elems per 16-row slab
  const unsigned long long gA =
      (unsigned long long)(uintptr_t)(Af + (size_t)bt * 2 * slabHalf);
  const uint32_t lds0 = (uint32_t)(uintptr_t)(&shA[0]);

  // ---- first slab via TDM: one wave-level tensor_load_to_lds ----
  if (wv == 0) {
    const unsigned int n4 = (unsigned int)(KT * 256);  // dwords in slab 0
    u32x4 g0;
    g0[0] = 1u;                                        // count=1, user mode
    g0[1] = lds0;                                      // lds_addr (bytes)
    g0[2] = (unsigned int)gA;                          // global_addr[31:0]
    g0[3] = ((unsigned int)(gA >> 32) & 0x01FFFFFFu) | (2u << 30);  // addr[56:32] | type=2
    i32x8 g1;
    g1[0] = (int)(2u << 16);                           // data_size=4B
    g1[1] = (int)((n4 & 0xFFFFu) << 16);               // tensor_dim0[15:0]
    g1[2] = (int)((n4 >> 16) & 0xFFFFu);               // tensor_dim0[31:16], tensor_dim1=0
    g1[3] = (int)((n4 & 0xFFFFu) << 16);               // tile_dim0 = n4
    g1[4] = 0;                                         // tile_dim1/2 unused
    g1[5] = (int)n4;                                   // tensor_dim0_stride low
    g1[6] = 0;
    g1[7] = 0;
    i32x4 gz4 = {0, 0, 0, 0};
    i32x8 gz8 = {0, 0, 0, 0, 0, 0, 0, 0};
    __builtin_amdgcn_tensor_load_to_lds(g0, g1, gz4, gz4, gz8, 0);
  }
  // ---- second slab via async global->LDS copies ----
  const int nchunk = KT * 64;  // 16-byte chunks in slab 1
  for (int c = tid; c < nchunk; c += 256) {
    uint32_t lds = lds0 + (uint32_t)(slabHalf * 2) + (uint32_t)c * 16u;
    unsigned long long ga = gA + (unsigned long long)slabHalf * 2 +
                            (unsigned long long)c * 16u;
    asm volatile("global_load_async_to_lds_b128 %0, %1, off"
                 :: "v"(lds), "v"(ga) : "memory");
  }
  __builtin_amdgcn_s_wait_tensorcnt(0);
  asm volatile("s_wait_asynccnt 0" ::: "memory");
  __syncthreads();

  v8f a0 = {}, a1 = {}, a2 = {}, a3 = {};   // rows [bt*32 .. +16)
  v8f c0 = {}, c1 = {}, c2 = {}, c3 = {};   // rows [bt*32+16 .. +32)
  const int nt0 = wv << 2;
  for (int kt = 0; kt < KT; ++kt) {
    const v16h alo = *(const v16h*)(&shA[(kt * 32 + lane) * 16]);
    const v16h ahi = *(const v16h*)(&shA[slabHalf + (kt * 32 + lane) * 16]);
    const _Float16* wp = Wf + (((size_t)kt * NT_ + nt0) * 32 + lane) * 16;
    const v16h b0 = *(const v16h*)(wp);
    const v16h b1 = *(const v16h*)(wp + 512);
    const v16h b2 = *(const v16h*)(wp + 1024);
    const v16h b3 = *(const v16h*)(wp + 1536);
    a0 = __builtin_amdgcn_wmma_f32_16x16x32_f16(false, alo, false, b0, (short)0, a0, false, false);
    a1 = __builtin_amdgcn_wmma_f32_16x16x32_f16(false, alo, false, b1, (short)0, a1, false, false);
    a2 = __builtin_amdgcn_wmma_f32_16x16x32_f16(false, alo, false, b2, (short)0, a2, false, false);
    a3 = __builtin_amdgcn_wmma_f32_16x16x32_f16(false, alo, false, b3, (short)0, a3, false, false);
    c0 = __builtin_amdgcn_wmma_f32_16x16x32_f16(false, ahi, false, b0, (short)0, c0, false, false);
    c1 = __builtin_amdgcn_wmma_f32_16x16x32_f16(false, ahi, false, b1, (short)0, c1, false, false);
    c2 = __builtin_amdgcn_wmma_f32_16x16x32_f16(false, ahi, false, b2, (short)0, c2, false, false);
    c3 = __builtin_amdgcn_wmma_f32_16x16x32_f16(false, ahi, false, b3, (short)0, c3, false, false);
  }
  // C/D layout: lane L: col=L&15; vgpr r: row = r + (L&16 ? 8 : 0)
  const int col = lane & 15;
  const int rsub = (lane & 16) ? 8 : 0;
  float* cb0 = Cout + (size_t)(bt * 32 + rsub) * NCOL_ + (nt0 << 4) + col;
  float* cb1 = cb0 + (size_t)16 * NCOL_;
#pragma unroll
  for (int r = 0; r < 8; ++r) {
    cb0[(size_t)r * NCOL_ + 0]  = a0[r];
    cb0[(size_t)r * NCOL_ + 16] = a1[r];
    cb0[(size_t)r * NCOL_ + 32] = a2[r];
    cb0[(size_t)r * NCOL_ + 48] = a3[r];
    cb1[(size_t)r * NCOL_ + 0]  = c0[r];
    cb1[(size_t)r * NCOL_ + 16] = c1[r];
    cb1[(size_t)r * NCOL_ + 32] = c2[r];
    cb1[(size_t)r * NCOL_ + 48] = c3[r];
  }
}

// ---------------- fused LSTM cell pointwise update ----------------
__global__ void k_cell(const float* __restrict__ pre, float* __restrict__ Hst,
                       float* __restrict__ Cst, const float* __restrict__ wc,
                       const float* __restrict__ bs, float* __restrict__ outC, int t) {
  int tid = blockIdx.x * blockDim.x + threadIdx.x;
  if (tid >= M_ * H_) return;
  int m = tid >> 7, h = tid & 127;
  const float* p = pre + (size_t)m * NCOL_;
  float cs = Cst[tid];
  float ig = sigm(p[h]            + bs[h]            + wc[h] * cs);
  float fg = sigm(p[H_ + h]       + bs[H_ + h]       + wc[H_ + h] * cs);
  float tt = tanhf(p[2 * H_ + h]  + bs[2 * H_ + h]);
  float cn = fg * cs + ig * tt;
  float og = sigm(p[3 * H_ + h]   + bs[3 * H_ + h]   + wc[2 * H_ + h] * cn);
  Cst[tid] = cn;
  Hst[tid] = og * tanhf(cn);
  if (outC) {
    int b = m >> 13, n = m & (N_ - 1);
    outC[(((size_t)(b * T_ + t)) * N_ + n) * H_ + h] = cn;
  }
}

// ---------------- host orchestration ----------------
static inline int cdiv(int a, int b) { return (a + b - 1) / b; }

extern "C" void kernel_launch(void* const* d_in, const int* in_sizes, int n_in,
                              void* d_out, int out_size, void* d_ws, size_t ws_size,
                              hipStream_t stream) {
  (void)in_sizes; (void)n_in; (void)out_size; (void)ws_size;
  const float* input = (const float*)d_in[0];
  const int*   ei    = (const int*)d_in[1];
  const float* ea    = (const float*)d_in[2];
  const float* mu    = (const float*)d_in[3];
  const float* lsc   = (const float*)d_in[4];
  const float* Wx0 = (const float*)d_in[5];
  const float* bx0 = (const float*)d_in[6];
  const float* Wh0 = (const float*)d_in[7];
  const float* bh0 = (const float*)d_in[8];
  const float* wc0 = (const float*)d_in[9];
  const float* bg0 = (const float*)d_in[10];
  const float* Wx1 = (const float*)d_in[11];
  const float* bx1 = (const float*)d_in[12];
  const float* Wh1 = (const float*)d_in[13];
  const float* bh1 = (const float*)d_in[14];
  const float* wc1 = (const float*)d_in[15];
  const float* bg1 = (const float*)d_in[16];
  float* out = (float*)d_out;

  // workspace carve (all fully re-initialized each call -> deterministic)
  uint8_t* w = (uint8_t*)d_ws;
  auto take = [&](size_t bytes) -> void* {
    void* p = (void*)w; w += (bytes + 255) & ~(size_t)255; return p;
  };
  float* wn   = (float*)take((size_t)E_ * 4);
  float* deg  = (float*)take((size_t)N_ * 4);
  float* xbuf = (float*)take((size_t)M_ * CIN_ * 4);
  float* H0 = (float*)take((size_t)M_ * H_ * 4);
  float* C0 = (float*)take((size_t)M_ * H_ * 4);
  float* H1 = (float*)take((size_t)M_ * H_ * 4);
  float* C1 = (float*)take((size_t)M_ * H_ * 4);
  float* P1  = (float*)take((size_t)M_ * H_ * 4);
  float* P2  = (float*)take((size_t)M_ * H_ * 4);
  float* P1h = (float*)take((size_t)M_ * H_ * 4);
  float* P2h = (float*)take((size_t)M_ * H_ * 4);
  _Float16* Acat = (_Float16*)take((size_t)MT_ * KT1_ * 512 * 2);
  float* pre = (float*)take((size_t)M_ * NCOL_ * 4);
  _Float16* Wf0 = (_Float16*)take((size_t)KT0_ * NT_ * 512 * 2);
  _Float16* Wf1 = (_Float16*)take((size_t)KT1_ * NT_ * 512 * 2);
  float* bs0 = (float*)take(NCOL_ * 4);
  float* bs1 = (float*)take(NCOL_ * 4);

  auto zero = [&](float* p, int n) {
    k_zero<<<cdiv(n, 256), 256, 0, stream>>>(p, n);
  };
  auto prop = [&](const float* z, float* o, int F) {
    zero(o, M_ * F);
    int nt = E_ * (F >> 2);
    k_prop<<<cdiv(nt, 256), 256, 0, stream>>>(z, o, ei, wn, F);
  };

  // graph normalization (sym, lambda_max=2)
  zero(deg, N_);
  k_deg<<<cdiv(E_, 256), 256, 0, stream>>>(ei, ea, deg);
  k_dinv<<<cdiv(N_, 256), 256, 0, stream>>>(deg);  // deg -> d^{-1/2} in place
  k_wn<<<cdiv(E_, 256), 256, 0, stream>>>(ei, ea, deg, wn);

  // weights -> f16 WMMA fragments; fused bias sums
  k_prep_w<<<cdiv(KT0_ * NT_ * 512, 256), 256, 0, stream>>>(Wx0, Wh0, CIN_, Wf0, KT0_);
  k_prep_w<<<cdiv(KT1_ * NT_ * 512, 256), 256, 0, stream>>>(Wx1, Wh1, H_, Wf1, KT1_);
  k_prep_bias<<<cdiv(NCOL_, 256), 256, 0, stream>>>(bx0, bh0, bg0, bs0);
  k_prep_bias<<<cdiv(NCOL_, 256), 256, 0, stream>>>(bx1, bh1, bg1, bs1);

  zero(H0, M_ * H_); zero(C0, M_ * H_);
  zero(H1, M_ * H_); zero(C1, M_ * H_);

  for (int t = 0; t < T_; ++t) {
    // ---- layer 0: input xbuf (F=64), hidden H0 ----
    k_xslice<<<cdiv(M_ * CIN_, 256), 256, 0, stream>>>(input, mu, lsc, xbuf, t);
    prop(xbuf, P1, CIN_);  prop(P1, P2, CIN_);   // Chebyshev T1, T2 of x
    prop(H0,   P1h, H_);   prop(P1h, P2h, H_);   // Chebyshev T1, T2 of H0
    k_pack<<<cdiv(MT_ * KT0_ * 512, 256), 256, 0, stream>>>(
        xbuf, P1, P2, CIN_, H0, P1h, P2h, Acat, KT0_);
    k_gemm<<<MT_ / 2, 256, 0, stream>>>(Acat, Wf0, pre, KT0_);
    k_cell<<<cdiv(M_ * H_, 256), 256, 0, stream>>>(pre, H0, C0, wc0, bs0, nullptr, t);

    // ---- layer 1: input = updated H0, hidden H1 ----
    prop(H0, P1, H_);   prop(P1, P2, H_);
    prop(H1, P1h, H_);  prop(P1h, P2h, H_);
    k_pack<<<cdiv(MT_ * KT1_ * 512, 256), 256, 0, stream>>>(
        H0, P1, P2, H_, H1, P1h, P2h, Acat, KT1_);
    k_gemm<<<MT_ / 2, 256, 0, stream>>>(Acat, Wf1, pre, KT1_);
    k_cell<<<cdiv(M_ * H_, 256), 256, 0, stream>>>(pre, H1, C1, wc1, bs1, out, t);
  }
}
